// CondLaneHead_51118700757230
// MI455X (gfx1250) — matrix operand order repaired and checked by reading, not compiled, not run on hardware
//
#include <hip/hip_runtime.h>
#include <hip/hip_bf16.h>

typedef __attribute__((ext_vector_type(16))) _Float16 v16h;
typedef __attribute__((ext_vector_type(8)))  _Float16 v8h;
typedef __attribute__((ext_vector_type(8)))  float    v8f;
typedef __attribute__((ext_vector_type(4)))  float    v4f;

// ---------------- constants for this problem ----------------
// B=16, C=256, H=40, W=100; dim=134, N=4, kd=128, ncls=2, mdim=256, inc=258
#define BB   16
#define CC   256
#define HH   40
#define WW   100
#define HWSZ 4000            // H*W
#define PP   64000           // B*H*W
#define CPAD_FEAT 288        // 258 padded to mult of 32
#define CPAD_DIM  160        // 134 padded
#define CPAD_M    256        // 256 already aligned
#define KD        128

// =============================================================
// build feat = concat(coords, features), NHWC f16, zero-padded
// =============================================================
__global__ void build_feat_kernel(const float* __restrict__ feats,
                                  _Float16* __restrict__ out)
{
    size_t i = (size_t)blockIdx.x * blockDim.x + threadIdx.x;
    size_t tot = (size_t)PP * CPAD_FEAT;
    if (i >= tot) return;
    int c = (int)(i % CPAD_FEAT);
    size_t p = i / CPAD_FEAT;
    int w = (int)(p % WW);
    int hb = (int)(p / WW);
    int h = hb % HH;
    int b = hb / HH;
    float v = 0.f;
    if (c == 0)              v = -1.f + 2.f * (float)w / (float)(WW - 1);
    else if (c == 1)         v = -1.f + 2.f * (float)h / (float)(HH - 1);
    else if (c < CC + 2)     v = feats[(((size_t)b * CC + (c - 2)) * HH + h) * WW + w];
    out[i] = (_Float16)v;
}

// =============================================================
// pack OIHW fp32 weights -> [Mpad, KHW*Cpad] f16, zero-padded
// =============================================================
__global__ void pack_w_kernel(const float* __restrict__ wsrc, _Float16* __restrict__ dst,
                              int Cout, int Cin, int KHW, int Cpad, int Mpad)
{
    const int ldK = KHW * Cpad;
    size_t i = (size_t)blockIdx.x * blockDim.x + threadIdx.x;
    size_t tot = (size_t)Mpad * ldK;
    if (i >= tot) return;
    int k = (int)(i % ldK);
    int m = (int)(i / ldK);
    int g = k / Cpad;
    int c = k - g * Cpad;
    float v = 0.f;
    if (m < Cout && c < Cin)
        v = wsrc[((size_t)m * Cin + c) * KHW + g];
    dst[i] = (_Float16)v;
}

__global__ void pack_b_kernel(const float* __restrict__ bsrc, float* __restrict__ dst,
                              int Cout, int Mpad)
{
    int m = blockIdx.x * blockDim.x + threadIdx.x;
    if (m < Mpad) dst[m] = (m < Cout) ? bsrc[m] : 0.f;
}

// =============================================================
// Implicit-GEMM 3x3/1x1 conv via V_WMMA_F32_16X16X32_F16.
// One wave per (MT*16) x (NT*16) output macro-tile; A fragments are
// reused across NT column tiles, B fragments across MT row tiles.
//   in  : NHWC f16, [P, Cpad]          (Cpad multiple of 32)
//   wpk : [Mpad, ngroups*Cpad] f16
//   out : ACT==0 -> f16 [P,Mpad] (bias+relu)
//         ACT==1 -> f32 [P,Mpad] (bias+sigmoid)
//         ACT==2 -> f32 [P,Mpad] (bias only)
// =============================================================
template<int ACT, int MT, int NT>
__global__ __launch_bounds__(32)
void conv_wmma_kernel(const _Float16* __restrict__ in,
                      const _Float16* __restrict__ wpk,
                      const float* __restrict__ bias,
                      void* __restrict__ out,
                      int Cpad, int ngroups, int Mpad)
{
    const int lane = threadIdx.x;      // 0..31
    const int hv   = lane >> 4;        // lane-half 0/1
    const int lan  = lane & 15;
    const int nb = blockIdx.x;         // N block of NT*16 positions
    const int mb = blockIdx.y;         // M block of MT*16 rows
    const int ldK = ngroups * Cpad;

    // per-lane output positions for each N tile
    int pb_[NT], ph_[NT], pw_[NT], pidx[NT];
    #pragma unroll
    for (int j = 0; j < NT; ++j) {
        const int p = (nb * NT + j) * 16 + lan;
        pidx[j] = p;
        const int b = p / HWSZ;
        const int rem = p - b * HWSZ;
        ph_[j] = rem / WW;
        pw_[j] = rem - ph_[j] * WW;
        pb_[j] = b;
    }

    // A-matrix row base per M tile (row owned by this lane)
    const _Float16* __restrict__ arow[MT];
    #pragma unroll
    for (int i = 0; i < MT; ++i)
        arow[i] = wpk + (size_t)((mb * MT + i) * 16 + lan) * ldK;

    v8f zero = {};
    v8f acc[MT][NT];
    #pragma unroll
    for (int i = 0; i < MT; ++i)
        #pragma unroll
        for (int j = 0; j < NT; ++j)
            acc[i][j] = zero;

    for (int k0 = 0; k0 < ldK; k0 += 32) {
        // ---- A fragments (16x32 f16, 16 contiguous K per lane-half) ----
        union frag { v16h v; v8h h2[2]; };
        frag a[MT];
        #pragma unroll
        for (int i = 0; i < MT; ++i) {
            if (k0 + 32 < ldK) __builtin_prefetch(arow[i] + k0 + 32, 0, 0);
            a[i].h2[0] = *(const v8h*)(arow[i] + k0 + hv * 16);
            a[i].h2[1] = *(const v8h*)(arow[i] + k0 + hv * 16 + 8);
        }

        // ---- B fragments: 16 contiguous K per lane-half ----
        const int kk = k0 + hv * 16;
        const int g  = kk / Cpad;
        const int c0 = kk - g * Cpad;
        int dy = 0, dx = 0;
        if (ngroups == 9) { dy = g / 3 - 1; dx = (g - (dy + 1) * 3) - 1; }
        frag bf[NT];
        #pragma unroll
        for (int j = 0; j < NT; ++j) {
            const int hh2 = ph_[j] + dy;
            const int ww2 = pw_[j] + dx;
            if (hh2 >= 0 && hh2 < HH && ww2 >= 0 && ww2 < WW) {
                const _Float16* __restrict__ src =
                    in + ((size_t)(pb_[j] * HWSZ + hh2 * WW + ww2)) * Cpad + c0;
                bf[j].h2[0] = *(const v8h*)(src);
                bf[j].h2[1] = *(const v8h*)(src + 8);
            } else {
                v8h z = {};
                bf[j].h2[0] = z;
                bf[j].h2[1] = z;
            }
        }

        // ---- MT*NT WMMAs with full fragment reuse ----
        #pragma unroll
        for (int i = 0; i < MT; ++i)
            #pragma unroll
            for (int j = 0; j < NT; ++j)
                acc[i][j] = __builtin_amdgcn_wmma_f32_16x16x32_f16(
                                false, a[i].v, false, bf[j].v,
                                (short)0, acc[i][j], false, false);
    }

    // ---- epilogue: D VGPR e holds M = tile*16 + hv*8 + e, N = lan ----
    #pragma unroll
    for (int i = 0; i < MT; ++i) {
        const int mbase = (mb * MT + i) * 16 + hv * 8;
        #pragma unroll
        for (int j = 0; j < NT; ++j) {
            if (ACT == 0) {
                _Float16* o = (_Float16*)out + (size_t)pidx[j] * Mpad + mbase;
                v8h r;
                #pragma unroll
                for (int e = 0; e < 8; ++e) {
                    float v = acc[i][j][e] + bias[mbase + e];
                    v = v > 0.f ? v : 0.f;
                    r[e] = (_Float16)v;
                }
                *(v8h*)o = r;
            } else {
                float* o = (float*)out + (size_t)pidx[j] * Mpad + mbase;
                v4f r0, r1;
                #pragma unroll
                for (int e = 0; e < 8; ++e) {
                    float v = acc[i][j][e] + bias[mbase + e];
                    if (ACT == 1) v = 1.f / (1.f + __expf(-v));
                    if (e < 4) r0[e] = v; else r1[e - 4] = v;
                }
                *(v4f*)(o)     = r0;
                *(v4f*)(o + 4) = r1;
            }
        }
    }
}

// =============================================================
// per-(b,n): argmax + sum of iam_prob, select params row,
// normalized spatial pooling -> inst[bn, 134]
// iam: f32 [P, 16] (sigmoid already applied); f4: f16 [P, 160]
// NOTE: reference does params[pos] with pos in [0,HW) -> batch-0 rows.
// =============================================================
__global__ __launch_bounds__(256)
void inst_kernel(const float* __restrict__ iam,
                 const _Float16* __restrict__ f4,
                 float* __restrict__ out_maskp,
                 float* __restrict__ out_regp,
                 float* __restrict__ inst)
{
    const int bn = blockIdx.x;
    const int b = bn >> 2;
    const int n = bn & 3;
    const int tid = threadIdx.x;
    __shared__ float smax[256];
    __shared__ int   sidx[256];
    __shared__ float ssum[256];

    float mx = -1e30f; int mi = 0; float sm = 0.f;
    for (int l = tid; l < HWSZ; l += 256) {
        float v = iam[(size_t)(b * HWSZ + l) * 16 + n];
        sm += v;
        if (v > mx) { mx = v; mi = l; }
    }
    smax[tid] = mx; sidx[tid] = mi; ssum[tid] = sm;
    __syncthreads();
    for (int s = 128; s > 0; s >>= 1) {
        if (tid < s) {
            ssum[tid] += ssum[tid + s];
            if (smax[tid + s] > smax[tid] ||
                (smax[tid + s] == smax[tid] && sidx[tid + s] < sidx[tid])) {
                smax[tid] = smax[tid + s];
                sidx[tid] = sidx[tid + s];
            }
        }
        __syncthreads();
    }
    const int pos = sidx[0];
    const float inv = 1.f / fmaxf(ssum[0], 1e-6f);

    if (tid < 67) {
        out_maskp[bn * 67 + tid] = (float)f4[(size_t)pos * CPAD_DIM + tid];
        out_regp [bn * 67 + tid] = (float)f4[(size_t)pos * CPAD_DIM + 67 + tid];
    }

    for (int c = tid; c < 134; c += 256) {
        float a = 0.f;
        for (int l = 0; l < HWSZ; ++l) {
            a += iam[(size_t)(b * HWSZ + l) * 16 + n] *
                 (float)f4[(size_t)(b * HWSZ + l) * CPAD_DIM + c];
        }
        inst[bn * 134 + c] = a * inv;
    }
}

// =============================================================
// small FC heads: pred_logits (134->2) and pred_kernel (134->128)
// =============================================================
__global__ __launch_bounds__(128)
void heads_kernel(const float* __restrict__ inst,
                  const float* __restrict__ cls_w, const float* __restrict__ cls_b,
                  const float* __restrict__ mk_w,  const float* __restrict__ mk_b,
                  float* __restrict__ out_logits, float* __restrict__ pred_k)
{
    const int bn = blockIdx.x;
    const int j = threadIdx.x;
    __shared__ float si[134];
    for (int c = j; c < 134; c += 128) si[c] = inst[bn * 134 + c];
    __syncthreads();
    float a = mk_b[j];
    for (int c = 0; c < 134; ++c) a += si[c] * mk_w[j * 134 + c];
    pred_k[bn * KD + j] = a;
    if (j < 2) {
        float g = cls_b[j];
        for (int c = 0; c < 134; ++c) g += si[c] * cls_w[j * 134 + c];
        out_logits[bn * 2 + j] = g;
    }
}

// =============================================================
// pred_masks[bn, l] = sum_k pred_kernel[bn,k] * mf[b, l, k]
// =============================================================
__global__ __launch_bounds__(256)
void masks_kernel(const float* __restrict__ pred_k,
                  const float* __restrict__ mf,
                  float* __restrict__ out_masks)
{
    const int bn = blockIdx.y;
    const int b = bn >> 2;
    __shared__ float sk[KD];
    if (threadIdx.x < KD) sk[threadIdx.x] = pred_k[bn * KD + threadIdx.x];
    __syncthreads();
    const int l = blockIdx.x * 256 + threadIdx.x;
    if (l >= HWSZ) return;
    const float* __restrict__ row = mf + (size_t)(b * HWSZ + l) * KD;
    float a = 0.f;
    #pragma unroll 4
    for (int k = 0; k < KD; ++k) a += sk[k] * row[k];
    out_masks[(size_t)bn * HWSZ + l] = a;
}

// =============================================================
// host-side launch
// =============================================================
static inline size_t alignUp256(size_t x) { return (x + 255) & ~(size_t)255; }

template<int ACT, int MT, int NT>
static void launch_conv(const _Float16* in, const _Float16* wpk, const float* bias,
                        void* out, int Cpad, int ngroups, int Mpad, hipStream_t stream)
{
    dim3 grid(PP / (NT * 16), Mpad / (MT * 16));
    conv_wmma_kernel<ACT, MT, NT><<<grid, 32, 0, stream>>>(in, wpk, bias, out, Cpad, ngroups, Mpad);
}

extern "C" void kernel_launch(void* const* d_in, const int* in_sizes, int n_in,
                              void* d_out, int out_size, void* d_ws, size_t ws_size,
                              hipStream_t stream) {
    (void)in_sizes; (void)n_in; (void)out_size; (void)ws_size;
    const float* features = (const float*)d_in[0];
    const float* iw0 = (const float*)d_in[1];  const float* ib0 = (const float*)d_in[2];
    const float* iw1 = (const float*)d_in[3];  const float* ib1 = (const float*)d_in[4];
    const float* iw2 = (const float*)d_in[5];  const float* ib2 = (const float*)d_in[6];
    const float* iw3 = (const float*)d_in[7];  const float* ib3 = (const float*)d_in[8];
    const float* iam_w = (const float*)d_in[9];  const float* iam_b = (const float*)d_in[10];
    const float* cls_w = (const float*)d_in[11]; const float* cls_b = (const float*)d_in[12];
    const float* mk_w  = (const float*)d_in[13]; const float* mk_b  = (const float*)d_in[14];
    const float* mw0 = (const float*)d_in[15]; const float* mb0 = (const float*)d_in[16];
    const float* mw1 = (const float*)d_in[17]; const float* mb1 = (const float*)d_in[18];
    const float* mw2 = (const float*)d_in[19]; const float* mb2 = (const float*)d_in[20];
    const float* mw3 = (const float*)d_in[21]; const float* mb3 = (const float*)d_in[22];
    const float* pw  = (const float*)d_in[23]; const float* pb  = (const float*)d_in[24];

    // ---- workspace carve ----
    char* ws = (char*)d_ws;
    size_t cur = 0;
    auto carve = [&](size_t bytes) { char* p = ws + cur; cur = alignUp256(cur + bytes); return p; };

    _Float16* FEAT = (_Float16*)carve((size_t)PP * CPAD_FEAT * 2);
    _Float16* PING = (_Float16*)carve((size_t)PP * CPAD_M * 2);
    _Float16* PONG = (_Float16*)carve((size_t)PP * CPAD_M * 2);
    _Float16* F4   = (_Float16*)carve((size_t)PP * CPAD_DIM * 2);
    float*    IAM  = (float*)   carve((size_t)PP * 16 * 4);
    float*    MF   = (float*)   carve((size_t)PP * KD * 4);

    _Float16* AW0  = (_Float16*)carve((size_t)160 * 9 * CPAD_FEAT * 2);
    _Float16* AW1  = (_Float16*)carve((size_t)160 * 9 * CPAD_DIM * 2);
    _Float16* AW2  = (_Float16*)carve((size_t)160 * 9 * CPAD_DIM * 2);
    _Float16* AW3  = (_Float16*)carve((size_t)160 * 9 * CPAD_DIM * 2);
    _Float16* AIAM = (_Float16*)carve((size_t)16  * 9 * CPAD_DIM * 2);
    _Float16* AM0  = (_Float16*)carve((size_t)256 * 9 * CPAD_FEAT * 2);
    _Float16* AM1  = (_Float16*)carve((size_t)256 * 9 * CPAD_M * 2);
    _Float16* AM2  = (_Float16*)carve((size_t)256 * 9 * CPAD_M * 2);
    _Float16* AM3  = (_Float16*)carve((size_t)256 * 9 * CPAD_M * 2);
    _Float16* APW  = (_Float16*)carve((size_t)KD * CPAD_M * 2);

    float* B0  = (float*)carve(160 * 4);
    float* B1  = (float*)carve(160 * 4);
    float* B2  = (float*)carve(160 * 4);
    float* B3  = (float*)carve(160 * 4);
    float* BIA = (float*)carve(16 * 4);
    float* BM0 = (float*)carve(256 * 4);
    float* BM1 = (float*)carve(256 * 4);
    float* BM2 = (float*)carve(256 * 4);
    float* BM3 = (float*)carve(256 * 4);
    float* BPW = (float*)carve(KD * 4);

    float* INST  = (float*)carve((size_t)64 * 134 * 4);
    float* PREDK = (float*)carve((size_t)64 * KD * 4);

    // ---- output sections (return order) ----
    float* out_logits = (float*)d_out;                  // 64*2
    float* out_maskp  = out_logits + 64 * 2;            // 64*67
    float* out_regp   = out_maskp + 64 * 67;            // 64*67
    float* out_masks  = out_regp + 64 * 67;             // 64*4000

    // ---- pack weights/biases ----
    auto packw = [&](const float* src, _Float16* dst, int Cout, int Cin, int KHW, int Cpad, int Mpad) {
        size_t tot = (size_t)Mpad * KHW * Cpad;
        pack_w_kernel<<<dim3((unsigned)((tot + 255) / 256)), 256, 0, stream>>>(src, dst, Cout, Cin, KHW, Cpad, Mpad);
    };
    auto packb = [&](const float* src, float* dst, int Cout, int Mpad) {
        pack_b_kernel<<<dim3((Mpad + 63) / 64), 64, 0, stream>>>(src, dst, Cout, Mpad);
    };
    packw(iw0, AW0, 134, 258, 9, CPAD_FEAT, 160);  packb(ib0, B0, 134, 160);
    packw(iw1, AW1, 134, 134, 9, CPAD_DIM, 160);   packb(ib1, B1, 134, 160);
    packw(iw2, AW2, 134, 134, 9, CPAD_DIM, 160);   packb(ib2, B2, 134, 160);
    packw(iw3, AW3, 134, 134, 9, CPAD_DIM, 160);   packb(ib3, B3, 134, 160);
    packw(iam_w, AIAM, 4, 134, 9, CPAD_DIM, 16);   packb(iam_b, BIA, 4, 16);
    packw(mw0, AM0, 256, 258, 9, CPAD_FEAT, 256);  packb(mb0, BM0, 256, 256);
    packw(mw1, AM1, 256, 256, 9, CPAD_M, 256);     packb(mb1, BM1, 256, 256);
    packw(mw2, AM2, 256, 256, 9, CPAD_M, 256);     packb(mb2, BM2, 256, 256);
    packw(mw3, AM3, 256, 256, 9, CPAD_M, 256);     packb(mb3, BM3, 256, 256);
    packw(pw,  APW, 128, 256, 1, CPAD_M, 128);     packb(pb,  BPW, 128, 128);

    // ---- build feat (coords + features), NHWC f16 ----
    {
        size_t tot = (size_t)PP * CPAD_FEAT;
        build_feat_kernel<<<dim3((unsigned)((tot + 255) / 256)), 256, 0, stream>>>(features, FEAT);
    }

    // param branch: feat -> f (134ch, padded 160); MT=2 (32 rows), NT=2 (32 cols)
    launch_conv<0, 2, 2>(FEAT, AW0, B0, PING, CPAD_FEAT, 9, 160, stream);
    launch_conv<0, 2, 2>(PING, AW1, B1, PONG, CPAD_DIM, 9, 160, stream);
    launch_conv<0, 2, 2>(PONG, AW2, B2, PING, CPAD_DIM, 9, 160, stream);
    launch_conv<0, 2, 2>(PING, AW3, B3, F4,   CPAD_DIM, 9, 160, stream);
    // iam conv + sigmoid -> f32 [P,16]; MT=1
    launch_conv<1, 1, 2>(F4, AIAM, BIA, IAM, CPAD_DIM, 9, 16, stream);

    // argmax/sum/pool + param select
    inst_kernel<<<64, 256, 0, stream>>>(IAM, F4, out_maskp, out_regp, INST);
    // FC heads
    heads_kernel<<<64, 128, 0, stream>>>(INST, cls_w, cls_b, mk_w, mk_b, out_logits, PREDK);

    // mask branch: feat -> m (256ch) -> mf (128ch f32); MT=4 (64 rows)
    launch_conv<0, 4, 2>(FEAT, AM0, BM0, PING, CPAD_FEAT, 9, 256, stream);
    launch_conv<0, 4, 2>(PING, AM1, BM1, PONG, CPAD_M, 9, 256, stream);
    launch_conv<0, 4, 2>(PONG, AM2, BM2, PING, CPAD_M, 9, 256, stream);
    launch_conv<0, 4, 2>(PING, AM3, BM3, PONG, CPAD_M, 9, 256, stream);
    launch_conv<2, 4, 2>(PONG, APW, BPW, MF, CPAD_M, 1, 128, stream);

    // pred_masks
    masks_kernel<<<dim3((HWSZ + 255) / 256, 64), 256, 0, stream>>>(PREDK, MF, out_masks);
}
